// Attention_9328668967755
// MI455X (gfx1250) — matrix operand order repaired and checked by reading, not compiled, not run on hardware
//
#include <hip/hip_runtime.h>
#include <hip/hip_bf16.h>

// ---------------------------------------------------------------------------
// Attention (b=8, c=256, n=48*48=2304, heads=4, d=32) via CDNA5 bf16 WMMA.
// Flash-attention structure: sim never materialized.
// ---------------------------------------------------------------------------

typedef __attribute__((ext_vector_type(16))) __bf16 v16bf;
typedef __attribute__((ext_vector_type(8)))  __bf16 v8bf;
typedef __attribute__((ext_vector_type(8)))  float  v8f;
typedef __attribute__((ext_vector_type(16))) float  v16f;

#define HEADS  4
#define DIMH   32
#define N_TOK  2304
#define C_IN   256
#define HID    128
#define QKSCALE 0.17677669529663687f   // 32^-0.5

#define CAT16(lo, hi) __builtin_shufflevector((lo), (hi), \
    0,1,2,3,4,5,6,7,8,9,10,11,12,13,14,15)

// Native bf16 convert (lowers to gfx1250 bf16 cvt ALU ops).
__device__ __forceinline__ __bf16 tobf(float f) { return (__bf16)f; }

__device__ __forceinline__ v16bf cvt16(float4 r0, float4 r1, float4 r2, float4 r3) {
  v16f t;
  t[0]=r0.x;  t[1]=r0.y;  t[2]=r0.z;  t[3]=r0.w;
  t[4]=r1.x;  t[5]=r1.y;  t[6]=r1.z;  t[7]=r1.w;
  t[8]=r2.x;  t[9]=r2.y;  t[10]=r2.z; t[11]=r2.w;
  t[12]=r3.x; t[13]=r3.y; t[14]=r3.z; t[15]=r3.w;
  return __builtin_convertvector(t, v16bf);
}

// XOR shuffle inside 16-lane half-wave groups via ds_swizzle (group-of-32 mode:
// offset = xor<<10 | or<<5 | and, and=0x1f). Masks < 16 stay inside the group.
#define SWZ_XOR(v, m) __builtin_bit_cast(float, \
    __builtin_amdgcn_ds_swizzle(__builtin_bit_cast(int, (v)), (((m) << 10) | 0x1F)))

__device__ __forceinline__ float redmax16(float v) {
  v = fmaxf(v, SWZ_XOR(v, 1));
  v = fmaxf(v, SWZ_XOR(v, 2));
  v = fmaxf(v, SWZ_XOR(v, 4));
  v = fmaxf(v, SWZ_XOR(v, 8));
  return v;
}
__device__ __forceinline__ float redsum16(float v) {
  v += SWZ_XOR(v, 1);
  v += SWZ_XOR(v, 2);
  v += SWZ_XOR(v, 4);
  v += SWZ_XOR(v, 8);
  return v;
}

// ---------------------------------------------------------------------------
// Kernel 1: qkv = w_qkv[384,256] @ x[b,256,2304]  (bf16 WMMA, f32 accum).
// Q,K stored [b,h,n,32] bf16 (Q pre-scaled); V stored [b,h,32,n] bf16.
// One wave = one 16x16 output tile; K-loop of 8 x wmma 16x16x32.
// ---------------------------------------------------------------------------
__global__ __launch_bounds__(256) void attn_qkv_kernel(
    const float* __restrict__ x, const float* __restrict__ w,
    __bf16* __restrict__ Q, __bf16* __restrict__ K, __bf16* __restrict__ V) {
  const int lane = threadIdx.x & 31;
  const int g  = lane >> 4;      // lane-half
  const int ln = lane & 15;
  const int wslot = __builtin_amdgcn_readfirstlane(threadIdx.x >> 5);
  const long wave = (long)blockIdx.x * 8 + wslot;
  if (wave >= 8L * 3456) return;
  const int b  = (int)(wave / 3456);
  const int t  = (int)(wave % 3456);
  const int o0 = (t / 144) * 16;           // output channel block (0..383)
  const int n0 = (t % 144) * 16;           // pixel block
  const float* xb = x + (long)b * C_IN * N_TOK;

  v8f acc = {};
  for (int c0 = 0; c0 < C_IN; c0 += 32) {
    // A fragment: W tile 16(M=o) x 32(K=c). Lane ln = row; two 8-elem K runs.
    const float* wr = w + (long)(o0 + ln) * C_IN + c0;
    const float4* p1 = reinterpret_cast<const float4*>(wr + g * 8);
    const float4* p2 = reinterpret_cast<const float4*>(wr + 16 + g * 8);
    v16bf a = cvt16(p1[0], p1[1], p2[0], p2[1]);
    // B fragment: X tile 32(K=c) x 16(N=n). Lane = col; K strided over c.
    v16f xv;
#pragma unroll
    for (int j = 0; j < 16; ++j) {
      int c = c0 + (j & 8) * 2 + g * 8 + (j & 7);
      xv[j] = xb[(long)c * N_TOK + n0 + ln];
    }
    v16bf bm = __builtin_convertvector(xv, v16bf);
    acc = __builtin_amdgcn_wmma_f32_16x16x32_bf16(false, a, false, bm,
                                                  (short)0, acc, false, false);
  }

  // Store into attention-friendly layouts. o0 is wave-uniform (SGPR) and a
  // 16-row tile never straddles q/k/v boundaries or a 32-row head.
  const int n = n0 + ln;
  if (o0 < HID) {                                   // ---- Q (pre-scaled)
    const int h = o0 >> 5, dbase = o0 & 31;
    __bf16* dst = Q + (((long)b * HEADS + h) * N_TOK + n) * DIMH + dbase;
#pragma unroll
    for (int r = 0; r < 8; ++r) dst[r + g * 8] = tobf(acc[r] * QKSCALE);
  } else if (o0 < 2 * HID) {                        // ---- K
    const int oo = o0 - HID, h = oo >> 5, dbase = oo & 31;
    __bf16* dst = K + (((long)b * HEADS + h) * N_TOK + n) * DIMH + dbase;
#pragma unroll
    for (int r = 0; r < 8; ++r) dst[r + g * 8] = tobf(acc[r]);
  } else {                                          // ---- V (d-major)
    const int oo = o0 - 2 * HID, h = oo >> 5, dbase = oo & 31;
    __bf16* dst = V + (((long)b * HEADS + h) * DIMH + dbase) * N_TOK + n;
#pragma unroll
    for (int r = 0; r < 8; ++r) dst[(long)(r + g * 8) * N_TOK] = tobf(acc[r]);
  }
}

// ---------------------------------------------------------------------------
// Kernel 2: flash attention. One wave owns a 16-row query block of one (b,h);
// j-loop in steps of 32: 2 wmma (scores) + online softmax + LDS transpose of
// P into A-fragment layout + 2 wmma (output accumulation).
// Output written bf16 as [b, n, 128] for the projection GEMM.
// ---------------------------------------------------------------------------
__global__ __launch_bounds__(256) void attn_flash_kernel(
    const __bf16* __restrict__ Q, const __bf16* __restrict__ K,
    const __bf16* __restrict__ V, __bf16* __restrict__ O) {
  __shared__ float ptile[8][16 * 36];      // per-wave 16x32 f32 tile, padded
  const int lane = threadIdx.x & 31;
  const int g  = lane >> 4;
  const int ln = lane & 15;
  const int wslot = __builtin_amdgcn_readfirstlane(threadIdx.x >> 5);
  float* lds = ptile[wslot];
  const long wave = (long)blockIdx.x * 8 + wslot;
  if (wave >= 4608) return;
  const int bh = (int)(wave / 144);
  const int i0 = (int)(wave % 144) * 16;

  const __bf16* Qp = Q + ((long)bh * N_TOK + i0) * DIMH;
  const __bf16* Kp = K + (long)bh * N_TOK * DIMH;
  const __bf16* Vp = V + (long)bh * DIMH * N_TOK;

  // Q A-fragment: 16(M=i) x 32(K=d); one contiguous 32B run per lane (k = 16g + s).
  v16bf qa;
  {
    const v8bf* q1 = reinterpret_cast<const v8bf*>(Qp + (long)ln * DIMH + g * 16);
    const v8bf* q2 = reinterpret_cast<const v8bf*>(Qp + (long)ln * DIMH + g * 16 + 8);
    qa = CAT16(q1[0], q2[0]);
  }

  float rm[8], rl[8];
#pragma unroll
  for (int r = 0; r < 8; ++r) { rm[r] = -3.0e38f; rl[r] = 0.0f; }
  v8f oa = {}, ob = {};

  for (int j0 = 0; j0 < N_TOK; j0 += 32) {
    // K B-fragments: [32(K=d) x 16(N=j)]; lane = j col, 32B contiguous d run.
    const v8bf* k0 = reinterpret_cast<const v8bf*>(Kp + (long)(j0 + ln) * DIMH + g * 16);
    const v8bf* k1 = reinterpret_cast<const v8bf*>(Kp + (long)(j0 + 16 + ln) * DIMH + g * 16);
    v16bf kb0 = CAT16(k0[0], k0[1]);
    v16bf kb1 = CAT16(k1[0], k1[1]);

    v8f zero = {};
    v8f s0 = __builtin_amdgcn_wmma_f32_16x16x32_bf16(false, qa, false, kb0,
                                                     (short)0, zero, false, false);
    v8f s1 = __builtin_amdgcn_wmma_f32_16x16x32_bf16(false, qa, false, kb1,
                                                     (short)0, zero, false, false);

    if (j0 + 32 < N_TOK) {   // hint next tiles into cache (global_prefetch_b8)
      __builtin_prefetch(Kp + (long)(j0 + 32 + ln) * DIMH, 0, 1);
      __builtin_prefetch(Vp + (long)ln * N_TOK + j0 + 32, 0, 1);
    }

    // Online softmax over this 16x32 score slab; a row lives across one
    // 16-lane half (C/D layout), so row stats reduce with 16-wide XOR swizzles.
    float p0[8], p1[8];
#pragma unroll
    for (int r = 0; r < 8; ++r) {
      float mx   = redmax16(fmaxf(s0[r], s1[r]));
      float nm   = fmaxf(rm[r], mx);
      float corr = __expf(rm[r] - nm);
      rm[r] = nm;
      float e0 = __expf(s0[r] - nm);
      float e1 = __expf(s1[r] - nm);
      p0[r] = e0; p1[r] = e1;
      rl[r] = rl[r] * corr + redsum16(e0 + e1);
      oa[r] *= corr;
      ob[r] *= corr;
    }

    // D-layout -> A-layout transpose of P through LDS (wave-private tile).
#pragma unroll
    for (int r = 0; r < 8; ++r) {
      const int row = r + g * 8;
      lds[row * 36 + ln]      = p0[r];
      lds[row * 36 + 16 + ln] = p1[r];
    }
    asm volatile("s_wait_dscnt 0" ::: "memory");
    const float4* t1 = reinterpret_cast<const float4*>(&lds[ln * 36 + g * 16]);
    const float4* t2 = reinterpret_cast<const float4*>(&lds[ln * 36 + g * 16 + 8]);
    v16bf pa = cvt16(t1[0], t1[1], t2[0], t2[1]);

    // V B-fragments: [32(K=j) x 16(N=d)] from V[d][n]; 32B contiguous j run.
    const v8bf* v0 = reinterpret_cast<const v8bf*>(Vp + (long)ln * N_TOK + j0 + g * 16);
    const v8bf* v1 = reinterpret_cast<const v8bf*>(Vp + (long)(16 + ln) * N_TOK + j0 + g * 16);
    v16bf vb0 = CAT16(v0[0], v0[1]);
    v16bf vb1 = CAT16(v1[0], v1[1]);

    oa = __builtin_amdgcn_wmma_f32_16x16x32_bf16(false, pa, false, vb0,
                                                 (short)0, oa, false, false);
    ob = __builtin_amdgcn_wmma_f32_16x16x32_bf16(false, pa, false, vb1,
                                                 (short)0, ob, false, false);
  }

  // Normalize and store bf16 at [b, n, h*32 + d].
  const int b = bh >> 2, h = bh & 3;
#pragma unroll
  for (int r = 0; r < 8; ++r) {
    const float inv = 1.0f / rl[r];
    const int i = i0 + r + g * 8;
    __bf16* dst = O + ((long)b * N_TOK + i) * HID + h * DIMH;
    dst[ln]      = tobf(oa[r] * inv);
    dst[16 + ln] = tobf(ob[r] * inv);
  }
}

// ---------------------------------------------------------------------------
// Kernel 3: y = w_out[256,128] @ attn_out[b,128,n] + b_out  (bf16 WMMA).
// attn_out is stored [b, n, 128] so B fragments are contiguous per lane.
// ---------------------------------------------------------------------------
__global__ __launch_bounds__(256) void attn_proj_kernel(
    const __bf16* __restrict__ Ob, const float* __restrict__ w,
    const float* __restrict__ bias, float* __restrict__ y) {
  const int lane = threadIdx.x & 31;
  const int g  = lane >> 4;
  const int ln = lane & 15;
  const int wslot = __builtin_amdgcn_readfirstlane(threadIdx.x >> 5);
  const long wave = (long)blockIdx.x * 8 + wslot;
  if (wave >= 8L * 2304) return;
  const int b  = (int)(wave / 2304);
  const int t  = (int)(wave % 2304);
  const int o0 = (t / 144) * 16;
  const int n0 = (t % 144) * 16;

  v8f acc = {};
  for (int c0 = 0; c0 < HID; c0 += 32) {
    const float* wr = w + (long)(o0 + ln) * HID + c0;
    const float4* p1 = reinterpret_cast<const float4*>(wr + g * 16);
    const float4* p2 = reinterpret_cast<const float4*>(wr + g * 16 + 8);
    v16bf a = cvt16(p1[0], p1[1], p2[0], p2[1]);

    const v8bf* pb = reinterpret_cast<const v8bf*>(
        Ob + ((long)b * N_TOK + n0 + ln) * HID + c0 + g * 16);
    v16bf bm = CAT16(pb[0], pb[1]);

    acc = __builtin_amdgcn_wmma_f32_16x16x32_bf16(false, a, false, bm,
                                                  (short)0, acc, false, false);
  }
#pragma unroll
  for (int r = 0; r < 8; ++r) {
    const int o = o0 + r + g * 8;
    y[((long)b * C_IN + o) * N_TOK + n0 + ln] = acc[r] + bias[o];
  }
}

// ---------------------------------------------------------------------------
extern "C" void kernel_launch(void* const* d_in, const int* in_sizes, int n_in,
                              void* d_out, int out_size, void* d_ws, size_t ws_size,
                              hipStream_t stream) {
  (void)in_sizes; (void)n_in; (void)out_size; (void)ws_size;
  const float* x     = (const float*)d_in[0];   // [8,256,48,48]
  const float* w_qkv = (const float*)d_in[1];   // [384,256]
  const float* w_out = (const float*)d_in[2];   // [256,128]
  const float* b_out = (const float*)d_in[3];   // [256]
  float* y = (float*)d_out;                     // [8,256,48,48] f32

  const size_t tsz = (size_t)8 * HEADS * N_TOK * DIMH * sizeof(__bf16); // 4.5 MB
  char* ws = (char*)d_ws;
  __bf16* Q  = (__bf16*)(ws);
  __bf16* K  = (__bf16*)(ws + tsz);
  __bf16* V  = (__bf16*)(ws + 2 * tsz);
  __bf16* Ob = (__bf16*)(ws + 3 * tsz);         // [8, 2304, 128] bf16

  attn_qkv_kernel  <<<3456, 256, 0, stream>>>(x, w_qkv, Q, K, V);
  attn_flash_kernel<<<576,  256, 0, stream>>>(Q, K, V, Ob);
  attn_proj_kernel <<<2304, 256, 0, stream>>>(Ob, w_out, b_out, y);
}
